// GPTClosedLoop_5351529251148
// MI455X (gfx1250) — matrix operand order repaired and checked by reading, not compiled, not run on hardware
//
#include <hip/hip_runtime.h>
#include <hip/hip_bf16.h>

// ---------------- model dims (compile-time) ----------------
#define NLAYER 3
#define DEMB   128
#define TT     64
#define BB     8
#define ROWS   (BB * TT)   // 512
#define DHEAD  32
#define HIDN   32
#define NXS    5

typedef __attribute__((ext_vector_type(16))) _Float16 v16h;
typedef __attribute__((ext_vector_type(8)))  float    v8f;

union HFrag {
    v16h     v;
    uint4    q[2];
    _Float16 e[16];
};

// ---------------- WMMA fragment loaders (contiguous K order per lane half, used by A and B alike) ----------------
// A: 16x32 f16, row-major source. lanes0-15: M=lane, K=0..15
//                                 lanes16-31: M=lane-16, K=16..31
__device__ __forceinline__ v16h load_a_rm(const _Float16* A, int lda, int m0, int k0, int lane) {
    int m   = m0 + (lane & 15);
    int klo = (lane < 16) ? 0 : 16;
    const _Float16* p = A + (size_t)m * lda + (k0 + klo);
    HFrag f;
    f.q[0] = *(const uint4*)(p);        // K = klo .. klo+7
    f.q[1] = *(const uint4*)(p + 8);    // K = klo+8 .. klo+15
    return f.v;
}

// B: 32x16 f16 from [N][K] row-major weights (x @ W.T layout).
// lanes0-15: col N=lane, K=k0..k0+15 ; lanes16-31: col N=lane-16, K=k0+16..k0+31
__device__ __forceinline__ v16h load_b_nm(const _Float16* B, int ldb, int n0, int k0, int lane) {
    int n  = n0 + (lane & 15);
    int kb = k0 + ((lane < 16) ? 0 : 16);
    const _Float16* p = B + (size_t)n * ldb + kb;
    HFrag f;
    f.q[0] = *(const uint4*)(p);
    f.q[1] = *(const uint4*)(p + 8);
    return f.v;
}

// B from K-major (row-major [K][N]) source, e.g. V matrix in attention.
__device__ __forceinline__ v16h load_b_km(const _Float16* B, int ldb, int n0, int k0, int lane) {
    int n  = n0 + (lane & 15);
    int kb = k0 + ((lane < 16) ? 0 : 16);
    HFrag f;
#pragma unroll
    for (int j = 0; j < 16; ++j) f.e[j] = B[(size_t)(kb + j) * ldb + n];
    return f.v;
}

__device__ __forceinline__ v8f wmma32(v16h a, v16h b, v8f c) {
    return __builtin_amdgcn_wmma_f32_16x16x32_f16(false, a, false, b, (short)0, c, false, false);
}

__device__ __forceinline__ float wave_sum(float v) {
#pragma unroll
    for (int off = 16; off > 0; off >>= 1) v += __shfl_xor(v, off, 32);
    return v;
}
__device__ __forceinline__ float wave_max(float v) {
#pragma unroll
    for (int off = 16; off > 0; off >>= 1) v = fmaxf(v, __shfl_xor(v, off, 32));
    return v;
}

__device__ __forceinline__ float gelu_f(float x) {
    float x3 = x * x * x;
    return 0.5f * x * (1.f + tanhf(0.7978845608f * (x + 0.044715f * x3)));
}

// ---------------- tiled WMMA GEMM over 32 waves (fully specialized) ----------------
// C = [RESID? Cf : 0] + A(f16,row-major) * B(f16,[N][K]) + bias ; optional GELU; f16/f32 out.
// M = ROWS fixed. K = KTILES*32. Row-tiles with (t = row & 63) > tcur are skipped (causal).
// Fragments for 4 k-steps (16 b128 loads) are preloaded before the 4 WMMAs so loads overlap.
template <int KTILES, int NN, bool RESID, bool GELU, bool F16OUT>
__device__ __forceinline__ void gemm_wmma(const _Float16* __restrict__ A, int lda,
                                          const _Float16* __restrict__ B, int ldb,
                                          const float* __restrict__ bias,
                                          float* __restrict__ Cf, _Float16* __restrict__ Ch,
                                          int ldc, int tcur, int wave, int lane)
{
    constexpr int nt = NN >> 4;
    constexpr int ntiles = (ROWS >> 4) * nt;
    for (int tidx = wave; tidx < ntiles; tidx += 32) {
        int mt = tidx / nt;
        int m0 = mt << 4;
        int n0 = (tidx - mt * nt) << 4;
        if ((m0 & (TT - 1)) > tcur) continue;   // fully-future row tile
        int n  = n0 + (lane & 15);
        int mb = m0 + ((lane < 16) ? 0 : 8);

        float bi = bias[n];                      // issue early; overlaps with WMMAs
        v8f acc;
        if (RESID) {
#pragma unroll
            for (int r2 = 0; r2 < 8; ++r2) acc[r2] = Cf[(size_t)(mb + r2) * ldc + n];
        } else {
#pragma unroll
            for (int r2 = 0; r2 < 8; ++r2) acc[r2] = 0.f;
        }

#pragma unroll
        for (int kc = 0; kc < KTILES; kc += 4) {
            v16h af[4], bf[4];
#pragma unroll
            for (int j = 0; j < 4; ++j) {
                af[j] = load_a_rm(A, lda, m0, (kc + j) * 32, lane);
                bf[j] = load_b_nm(B, ldb, n0, (kc + j) * 32, lane);
            }
#pragma unroll
            for (int j = 0; j < 4; ++j) acc = wmma32(af[j], bf[j], acc);
        }

        if (F16OUT) {
            _Float16 ov[8];
#pragma unroll
            for (int r2 = 0; r2 < 8; ++r2) {
                float v = acc[r2] + bi;
                if (GELU) v = gelu_f(v);
                ov[r2] = (_Float16)v;
            }
#pragma unroll
            for (int r2 = 0; r2 < 8; ++r2)
                Ch[(size_t)(mb + r2) * ldc + n] = ov[r2];
        } else {
            float ov[8];
#pragma unroll
            for (int r2 = 0; r2 < 8; ++r2) ov[r2] = acc[r2] + bi;
#pragma unroll
            for (int r2 = 0; r2 < 8; ++r2)
                Cf[(size_t)(mb + r2) * ldc + n] = ov[r2];
        }
    }
}

// ---------------- LayerNorm (rows t<=tcur; zero-fill future rows for safety) ----------------
__device__ void layernorm_pass(const float* __restrict__ h,
                               const float* __restrict__ g, const float* __restrict__ be,
                               _Float16* __restrict__ x16, int tcur, int wave, int lane)
{
    for (int row = wave; row < ROWS; row += 32) {
        int t = row & (TT - 1);
        _Float16* xp = x16 + (size_t)row * DEMB;
        if (t > tcur) {
#pragma unroll
            for (int j = 0; j < 4; ++j) xp[lane + 32 * j] = (_Float16)0.f;
            continue;
        }
        const float* hp = h + (size_t)row * DEMB;
        float v[4];
        float s = 0.f, ss = 0.f;
#pragma unroll
        for (int j = 0; j < 4; ++j) {
            v[j] = hp[lane + 32 * j];
            s += v[j];
            ss += v[j] * v[j];
        }
        s  = wave_sum(s);
        ss = wave_sum(ss);
        float mean = s * (1.f / DEMB);
        float var  = ss * (1.f / DEMB) - mean * mean;
        float rstd = rsqrtf(var + 1e-5f);
#pragma unroll
        for (int j = 0; j < 4; ++j) {
            int c = lane + 32 * j;
            xp[c] = (_Float16)((v[j] - mean) * rstd * g[c] + be[c]);
        }
    }
}

// ---------------- per-wave causal attention: wave w owns (b, head) = (w>>2, w&3) --------------
__device__ void attention_wave(const _Float16* __restrict__ qkv16,
                               float* __restrict__ scbuf, _Float16* __restrict__ attbuf,
                               _Float16* __restrict__ obuf,
                               int tcur, int wave, int lane)
{
    int b  = wave >> 2;
    int hd = wave & 3;
    const _Float16* qb = qkv16 + (size_t)(b * TT) * (3 * DEMB) + hd * DHEAD;
    const _Float16* kb = qb + DEMB;
    const _Float16* vb = qb + 2 * DEMB;
    float*     sc  = scbuf  + (size_t)wave * TT * TT;
    _Float16*  att = attbuf + (size_t)wave * TT * TT;
    const float scale = 0.17677669529663687f;   // 1/sqrt(32)

    // scores = q @ k^T * scale  (lower-triangular tiles only)
    for (int m0 = 0; m0 <= tcur; m0 += 16) {
        v16h a = load_a_rm(qb, 3 * DEMB, m0, 0, lane);   // reused across the row of tiles
        for (int n0 = 0; n0 <= m0; n0 += 16) {
            v16h bf = load_b_nm(kb, 3 * DEMB, n0, 0, lane);
            v8f acc;
#pragma unroll
            for (int r2 = 0; r2 < 8; ++r2) acc[r2] = 0.f;
            acc = wmma32(a, bf, acc);
            int n  = n0 + (lane & 15);
            int mb = m0 + ((lane < 16) ? 0 : 8);
            float ov[8];
#pragma unroll
            for (int r2 = 0; r2 < 8; ++r2) ov[r2] = acc[r2] * scale;
#pragma unroll
            for (int r2 = 0; r2 < 8; ++r2)
                sc[(size_t)(mb + r2) * TT + n] = ov[r2];
        }
    }

    // causal softmax per row; write full 64 cols (zeros beyond r) as f16
    for (int r = 0; r <= tcur; ++r) {
        float s0 = (lane       <= r) ? sc[(size_t)r * TT + lane]      : -1e30f;
        float s1 = (lane + 32  <= r) ? sc[(size_t)r * TT + lane + 32] : -1e30f;
        float mx = wave_max(fmaxf(s0, s1));
        float e0 = (lane      <= r) ? __expf(s0 - mx) : 0.f;
        float e1 = (lane + 32 <= r) ? __expf(s1 - mx) : 0.f;
        float sm  = wave_sum(e0 + e1);
        float inv = 1.f / sm;
        att[(size_t)r * TT + lane]      = (_Float16)(e0 * inv);
        att[(size_t)r * TT + lane + 32] = (_Float16)(e1 * inv);
    }

    // o = att @ v  -> write into obuf rows [b*64 + m], cols [hd*32 + n]
    for (int m0 = 0; m0 <= tcur; m0 += 16) {
        for (int n0 = 0; n0 < DHEAD; n0 += 16) {
            v16h a0 = load_a_rm(att, TT, m0, 0, lane);
            v16h a1 = load_a_rm(att, TT, m0, 32, lane);
            v16h b0 = load_b_km(vb, 3 * DEMB, n0, 0, lane);
            v16h b1 = load_b_km(vb, 3 * DEMB, n0, 32, lane);
            v8f acc;
#pragma unroll
            for (int r2 = 0; r2 < 8; ++r2) acc[r2] = 0.f;
            acc = wmma32(a0, b0, acc);
            acc = wmma32(a1, b1, acc);
            int n  = n0 + (lane & 15);
            int mb = m0 + ((lane < 16) ? 0 : 8);
            _Float16 ov[8];
#pragma unroll
            for (int r2 = 0; r2 < 8; ++r2) ov[r2] = (_Float16)acc[r2];
#pragma unroll
            for (int r2 = 0; r2 < 8; ++r2)
                obuf[(size_t)(b * TT + mb + r2) * DEMB + hd * DHEAD + n] = ov[r2];
        }
    }
}

// ---------------- parameter bundle ----------------
struct Params {
    const float *r;
    const float *G1A, *G1B, *G1C, *G1D, *G2A, *G2B, *G2C, *G2D;
    const float *w1, *b1, *w2, *b2;
    const float *wte_w, *wte_b, *wpe;
    const float *ln1_w, *ln1_b, *qkv_b, *attn_proj_b, *ln2_w, *ln2_b;
    const float *fc_b, *mlp_proj_b, *lnf_w, *lnf_b, *head_w, *head_b;
    const _Float16 *qkvW, *projW, *fcW, *mlpW;   // converted f16 weights
    float    *h;
    _Float16 *x16, *qkv16, *u16;
    float    *sc;
    _Float16 *att16;
    float    *out;
};

// ---------------- one-shot f32 -> f16 weight conversion ----------------
__global__ void convert_w_kernel(const float* __restrict__ a, const float* __restrict__ b,
                                 const float* __restrict__ c, const float* __restrict__ d,
                                 _Float16* __restrict__ oa, _Float16* __restrict__ ob,
                                 _Float16* __restrict__ oc, _Float16* __restrict__ od)
{
    const int n1 = NLAYER * 3 * DEMB * DEMB;   // qkv_w
    const int n2 = NLAYER * DEMB * DEMB;       // attn_proj_w
    const int n3 = NLAYER * 4 * DEMB * DEMB;   // fc_w
    const int n4 = NLAYER * DEMB * 4 * DEMB;   // mlp_proj_w
    const int total = n1 + n2 + n3 + n4;
    for (int i = blockIdx.x * blockDim.x + threadIdx.x; i < total; i += gridDim.x * blockDim.x) {
        if (i < n1)                oa[i]                = (_Float16)a[i];
        else if (i < n1 + n2)      ob[i - n1]           = (_Float16)b[i - n1];
        else if (i < n1 + n2 + n3) oc[i - n1 - n2]      = (_Float16)c[i - n1 - n2];
        else                       od[i - n1 - n2 - n3] = (_Float16)d[i - n1 - n2 - n3];
    }
}

// ---------------- persistent single-workgroup closed-loop GPT ----------------
__global__ __launch_bounds__(1024) void gpt_closed_loop_kernel(Params p)
{
    __shared__ float Els[ROWS];            // E[b][t]
    __shared__ float Uls[ROWS];            // U[b][t]
    __shared__ float x1s[BB * NXS];
    __shared__ float x2s[BB * NXS];
    __shared__ float yv[BB];
    __shared__ float uvec[BB];

    const int tid  = threadIdx.x;
    const int lane = tid & 31;
    const int wave = tid >> 5;

    for (int idx = tid; idx < ROWS; idx += 1024) { Els[idx] = 0.f; Uls[idx] = 0.f; }
    if (tid < BB * NXS) { x1s[tid] = 0.f; x2s[tid] = 0.f; }
    if (tid < BB) yv[tid] = 0.f;
    __syncthreads();

    for (int i = 0; i < TT; ++i) {
        // Y[:, i] = incoming y ; E[:, i] = r[:, i] - y
        if (tid < BB) {
            p.out[tid * TT + i] = yv[tid];
            Els[tid * TT + i]   = p.r[tid * TT + i] - yv[tid];
        }
        __syncthreads();

        // embedding: h = [E,U] @ wte_w.T + wte_b + wpe   (zero rows with t > i)
        for (int idx = tid; idx < ROWS * DEMB; idx += 1024) {
            int row = idx >> 7;
            int c   = idx & 127;
            int t   = row & (TT - 1);
            int b   = row >> 6;
            float val = 0.f;
            if (t <= i) {
                val = Els[b * TT + t] * p.wte_w[2 * c] + Uls[b * TT + t] * p.wte_w[2 * c + 1]
                    + p.wte_b[c] + p.wpe[t * DEMB + c];
            }
            p.h[idx] = val;
        }
        __syncthreads();

        for (int L = 0; L < NLAYER; ++L) {
            layernorm_pass(p.h, p.ln1_w + L * DEMB, p.ln1_b + L * DEMB, p.x16, i, wave, lane);
            __syncthreads();
            gemm_wmma<4, 3 * DEMB, false, false, true>(
                p.x16, DEMB, p.qkvW + (size_t)L * 3 * DEMB * DEMB, DEMB,
                p.qkv_b + L * 3 * DEMB, nullptr, p.qkv16, 3 * DEMB, i, wave, lane);
            __syncthreads();
            attention_wave(p.qkv16, p.sc, p.att16, p.x16, i, wave, lane);
            __syncthreads();
            gemm_wmma<4, DEMB, true, false, false>(
                p.x16, DEMB, p.projW + (size_t)L * DEMB * DEMB, DEMB,
                p.attn_proj_b + L * DEMB, p.h, nullptr, DEMB, i, wave, lane);
            __syncthreads();
            layernorm_pass(p.h, p.ln2_w + L * DEMB, p.ln2_b + L * DEMB, p.x16, i, wave, lane);
            __syncthreads();
            gemm_wmma<4, 4 * DEMB, false, true, true>(
                p.x16, DEMB, p.fcW + (size_t)L * 4 * DEMB * DEMB, DEMB,
                p.fc_b + L * 4 * DEMB, nullptr, p.u16, 4 * DEMB, i, wave, lane);
            __syncthreads();
            gemm_wmma<16, DEMB, true, false, false>(
                p.u16, 4 * DEMB, p.mlpW + (size_t)L * DEMB * 4 * DEMB, 4 * DEMB,
                p.mlp_proj_b + L * DEMB, p.h, nullptr, DEMB, i, wave, lane);
            __syncthreads();
        }

        layernorm_pass(p.h, p.lnf_w, p.lnf_b, p.x16, i, wave, lane);
        __syncthreads();

        // head: u_next[b] = lnf(h)[b, i] . head_w + head_b  (waves 0..7 -> one b each)
        if (wave < BB) {
            int row = wave * TT + i;
            float s = 0.f;
#pragma unroll
            for (int j = 0; j < 4; ++j) {
                int c = lane + 32 * j;
                s += (float)p.x16[(size_t)row * DEMB + c] * p.head_w[c];
            }
            s = wave_sum(s);
            if (lane == 0) uvec[wave] = s + p.head_b[0];
        }
        __syncthreads();

        if (tid < BB && i + 1 < TT) Uls[tid * TT + i + 1] = uvec[tid];

        // tiny SSM update: wave 0, lane b handles batch b
        if (wave == 0 && lane < BB) {
            int b = lane;
            float u0 = Uls[i];        // U[0, i, 0]
            float u1 = uvec[0];       // u_next[0, 0]
            float x1v[NXS], x1a[NXS], x1n[NXS], x2v[NXS], x2n[NXS];
#pragma unroll
            for (int j = 0; j < NXS; ++j) { x1v[j] = x1s[b * NXS + j]; x2v[j] = x2s[b * NXS + j]; }
#pragma unroll
            for (int j = 0; j < NXS; ++j) {
                float s = p.G1B[b * NXS + j] * u0;
#pragma unroll
                for (int k = 0; k < NXS; ++k) s += p.G1A[(b * NXS + j) * NXS + k] * x1v[k];
                x1a[j] = s;
            }
            float y1 = p.G1D[b] * u1;
#pragma unroll
            for (int j = 0; j < NXS; ++j) y1 += p.G1C[b * NXS + j] * x1a[j];
#pragma unroll
            for (int j = 0; j < NXS; ++j) {
                float s = p.G1B[b * NXS + j] * u1;
#pragma unroll
                for (int k = 0; k < NXS; ++k) s += p.G1A[(b * NXS + j) * NXS + k] * x1a[k];
                x1n[j] = s;
            }
            float acc2 = p.b2[0];
#pragma unroll
            for (int hh = 0; hh < HIDN; ++hh)
                acc2 += tanhf(y1 * p.w1[hh] + p.b1[hh]) * p.w2[hh];
            float y2 = acc2;
            float y3 = p.G2D[b] * y2;
#pragma unroll
            for (int j = 0; j < NXS; ++j) y3 += p.G2C[b * NXS + j] * x2v[j];
#pragma unroll
            for (int j = 0; j < NXS; ++j) {
                float s = p.G2B[b * NXS + j] * y2;
#pragma unroll
                for (int k = 0; k < NXS; ++k) s += p.G2A[(b * NXS + j) * NXS + k] * x2v[k];
                x2n[j] = s;
            }
#pragma unroll
            for (int j = 0; j < NXS; ++j) { x1s[b * NXS + j] = x1n[j]; x2s[b * NXS + j] = x2n[j]; }
            yv[b] = y3;
        }
        __syncthreads();
    }
}

// ---------------- host-side launch ----------------
extern "C" void kernel_launch(void* const* d_in, const int* in_sizes, int n_in,
                              void* d_out, int out_size, void* d_ws, size_t ws_size,
                              hipStream_t stream)
{
    (void)in_sizes; (void)n_in; (void)out_size; (void)ws_size;

    const float* r           = (const float*)d_in[0];
    const float* G1A         = (const float*)d_in[1];
    const float* G1B         = (const float*)d_in[2];
    const float* G1C         = (const float*)d_in[3];
    const float* G1D         = (const float*)d_in[4];
    const float* G2A         = (const float*)d_in[5];
    const float* G2B         = (const float*)d_in[6];
    const float* G2C         = (const float*)d_in[7];
    const float* G2D         = (const float*)d_in[8];
    const float* w1          = (const float*)d_in[9];
    const float* b1          = (const float*)d_in[10];
    const float* w2          = (const float*)d_in[11];
    const float* b2          = (const float*)d_in[12];
    const float* wte_w       = (const float*)d_in[13];
    const float* wte_b       = (const float*)d_in[14];
    const float* wpe         = (const float*)d_in[15];
    const float* ln1_w       = (const float*)d_in[16];
    const float* ln1_b       = (const float*)d_in[17];
    const float* qkv_w       = (const float*)d_in[18];
    const float* qkv_b       = (const float*)d_in[19];
    const float* attn_proj_w = (const float*)d_in[20];
    const float* attn_proj_b = (const float*)d_in[21];
    const float* ln2_w       = (const float*)d_in[22];
    const float* ln2_b       = (const float*)d_in[23];
    const float* fc_w        = (const float*)d_in[24];
    const float* fc_b        = (const float*)d_in[25];
    const float* mlp_proj_w  = (const float*)d_in[26];
    const float* mlp_proj_b  = (const float*)d_in[27];
    const float* lnf_w       = (const float*)d_in[28];
    const float* lnf_b       = (const float*)d_in[29];
    const float* head_w      = (const float*)d_in[30];
    const float* head_b      = (const float*)d_in[31];
    // d_in[32] = n_head (constant 4, baked in)

    char* base = (char*)d_ws;
    size_t off = 0;
    auto take = [&](size_t bytes) -> void* {
        void* ptr = base + off;
        off = (off + bytes + 255) & ~(size_t)255;
        return ptr;
    };

    _Float16* qkvW  = (_Float16*)take((size_t)NLAYER * 3 * DEMB * DEMB * sizeof(_Float16));
    _Float16* projW = (_Float16*)take((size_t)NLAYER * DEMB * DEMB * sizeof(_Float16));
    _Float16* fcW   = (_Float16*)take((size_t)NLAYER * 4 * DEMB * DEMB * sizeof(_Float16));
    _Float16* mlpW  = (_Float16*)take((size_t)NLAYER * DEMB * 4 * DEMB * sizeof(_Float16));
    float*    h     = (float*)   take((size_t)ROWS * DEMB * sizeof(float));
    _Float16* x16   = (_Float16*)take((size_t)ROWS * DEMB * sizeof(_Float16));
    _Float16* qkv16 = (_Float16*)take((size_t)ROWS * 3 * DEMB * sizeof(_Float16));
    _Float16* u16   = (_Float16*)take((size_t)ROWS * 4 * DEMB * sizeof(_Float16));
    float*    sc    = (float*)   take((size_t)32 * TT * TT * sizeof(float));
    _Float16* att16 = (_Float16*)take((size_t)32 * TT * TT * sizeof(_Float16));

    convert_w_kernel<<<dim3(512), dim3(256), 0, stream>>>(
        qkv_w, attn_proj_w, fc_w, mlp_proj_w, qkvW, projW, fcW, mlpW);

    Params p;
    p.r = r;
    p.G1A = G1A; p.G1B = G1B; p.G1C = G1C; p.G1D = G1D;
    p.G2A = G2A; p.G2B = G2B; p.G2C = G2C; p.G2D = G2D;
    p.w1 = w1; p.b1 = b1; p.w2 = w2; p.b2 = b2;
    p.wte_w = wte_w; p.wte_b = wte_b; p.wpe = wpe;
    p.ln1_w = ln1_w; p.ln1_b = ln1_b;
    p.qkv_b = qkv_b; p.attn_proj_b = attn_proj_b;
    p.ln2_w = ln2_w; p.ln2_b = ln2_b;
    p.fc_b = fc_b; p.mlp_proj_b = mlp_proj_b;
    p.lnf_w = lnf_w; p.lnf_b = lnf_b;
    p.head_w = head_w; p.head_b = head_b;
    p.qkvW = qkvW; p.projW = projW; p.fcW = fcW; p.mlpW = mlpW;
    p.h = h; p.x16 = x16; p.qkv16 = qkv16; p.u16 = u16;
    p.sc = sc; p.att16 = att16;
    p.out = (float*)d_out;

    gpt_closed_loop_kernel<<<dim3(1), dim3(1024), 0, stream>>>(p);
}
